// CMHAttention_88338887344217
// MI455X (gfx1250) — matrix operand order repaired and checked by reading, not compiled, not run on hardware
//
#include <hip/hip_runtime.h>

typedef __bf16 bf16_t;
typedef __attribute__((ext_vector_type(8)))  __bf16 v8bf;
typedef __attribute__((ext_vector_type(16))) __bf16 v16bf;
typedef __attribute__((ext_vector_type(8)))  float  v8f;
typedef __attribute__((ext_vector_type(4)))  unsigned int u32x4;
typedef __attribute__((ext_vector_type(8)))  int i32x8;
typedef __attribute__((ext_vector_type(4)))  int i32x4;

// ---------------------------------------------------------------- helpers
__device__ __forceinline__ v16bf cat8(v8bf lo, v8bf hi) {
    return __builtin_shufflevector(lo, hi, 0,1,2,3,4,5,6,7,8,9,10,11,12,13,14,15);
}

__device__ __forceinline__ v8f wmma_bf16(v16bf a, v16bf b, v8f c) {
    return __builtin_amdgcn_wmma_f32_16x16x32_bf16(false, a, false, b, (short)0, c, false, false);
}

// A-fragment (16x32 bf16): elems 0..7 <-> k=kc+lh*8+e, elems 8..15 <-> k=kc+16+lh*8+(e-8)
__device__ __forceinline__ v16bf load_afrag(const bf16_t* tile, int row, int kc,
                                            int lm, int lh, int ld) {
    const bf16_t* p = tile + (row + lm) * ld + kc + lh * 8;
    return cat8(*(const v8bf*)p, *(const v8bf*)(p + 16));
}

// B-fragment (Bt stored as [N][K] rows): elems 0..7 <-> k=kc+lh*8+e, elems 8..15 <-> k=kc+16+lh*8+(e-8)
__device__ __forceinline__ v16bf load_bfrag(const bf16_t* tile, int row, int kc,
                                            int lm, int lh, int ld) {
    const bf16_t* p = tile + (row + lm) * ld + kc + lh * 8;
    return cat8(*(const v8bf*)p, *(const v8bf*)(p + 16));
}

// ---- TDM: 2-D tile (rows x rowLen bf16 elems) global -> LDS with row padding.
// padIntCode: pad after 2^(code+1) dwords; padAmtCode: (code+1) dwords of pad.
__device__ __forceinline__ void tdm_load_2d(unsigned ldsOff, const void* gptr,
                                            unsigned rowLen, unsigned rows,
                                            unsigned rowStride,
                                            unsigned padIntCode, unsigned padAmtCode) {
    unsigned long long ga = (unsigned long long)gptr;
    u32x4 g0;
    g0[0] = 1u;                                             // count=1, user mode
    g0[1] = ldsOff;                                         // lds_addr (bytes)
    g0[2] = (unsigned)(ga & 0xffffffffu);                   // global_addr lo
    g0[3] = (unsigned)((ga >> 32) & 0x01ffffffu) | (2u << 30); // addr[56:32] | type=2
    i32x8 g1;
    g1[0] = (int)((1u << 16)                                // data_size = 2 bytes
                | (1u << 20)                                // pad_enable
                | (padIntCode << 22)
                | (padAmtCode << 25));
    g1[1] = (int)((rowLen & 0xffffu) << 16);                // tensor_dim0[15:0]
    g1[2] = (int)(((rowLen >> 16) & 0xffffu) | ((rows & 0xffffu) << 16)); // td0 hi | td1 lo
    g1[3] = (int)(((rows >> 16) & 0xffffu) | ((rowLen & 0xffffu) << 16)); // td1 hi | tile_dim0
    g1[4] = (int)(rows & 0xffffu);                          // tile_dim1 (tile_dim2 = 0)
    g1[5] = (int)rowStride;                                 // tensor_dim0_stride lo32
    g1[6] = 0;                                              // stride hi | td1_stride lo
    g1[7] = 0;
    i32x4 z4 = {0, 0, 0, 0};
    i32x8 z8 = {0, 0, 0, 0, 0, 0, 0, 0};
    __builtin_amdgcn_tensor_load_to_lds(g0, g1, z4, z4, z8, 0);
}

// ---------------------------------------------------------------- fp32 -> bf16
__global__ __launch_bounds__(256) void cvt_f32_bf16(const float* __restrict__ in,
                                                    bf16_t* __restrict__ out, long n) {
    long i = (long)blockIdx.x * 256 + threadIdx.x;
    long stride = (long)gridDim.x * 256;
    for (; i < n; i += stride) out[i] = (bf16_t)in[i];
}

// ---------------------------------------------------------------- generic GEMM
// C[M,N] = A[M,K] * B[N,K]^T (row-major bf16), f32 accumulate.
// Double-buffered LDS staging via GLOBAL_LOAD_ASYNC_TO_LDS_B128 (ASYNCcnt).
// OM: 0 = bf16 normal, 1 = bf16 transposed (O[n][m]), 2 = f32 + bias.
template<int TBM, int TBN, int WGM, int WGN, int OM>
__global__ __launch_bounds__(256) void gemm_abt(
    const bf16_t* __restrict__ Ab, const bf16_t* __restrict__ Bb,
    void* __restrict__ Ob, const float* __restrict__ bias,
    int Kdim, int lda, int ldb, int ldo,
    long aStr, long bStr, long oStr, int aMode, int bMode, int Hmod)
{
    constexpr int BK = 64;
    constexpr int LROW = BK + 8;                 // padded LDS row stride (halves)
    constexpr int ABYTES = TBM * LROW * 2;
    constexpr int BBYTES = TBN * LROW * 2;
    constexpr int BUFBYTES = ABYTES + BBYTES;
    constexpr int WM = TBM / WGM, WN = TBN / WGN;
    constexpr int FM = WM / 16,  FN = WN / 16;
    constexpr int G = TBM / 32 + TBN / 32;       // async instrs per wave per stage

    extern __shared__ char smem[];

    const int z = blockIdx.z;
    auto sel = [&](int mode) -> long {
        return mode == 0 ? (long)z : (mode == 1 ? (long)(z / Hmod) : (long)(z % Hmod));
    };
    const bf16_t* A  = Ab + sel(aMode) * aStr;
    const bf16_t* Bm = Bb + sel(bMode) * bStr;

    const int m0 = blockIdx.y * TBM;
    const int n0 = blockIdx.x * TBN;
    const int t = threadIdx.x;
    const int w = t >> 5, lane = t & 31;
    const int wm = w / WGN, wn = w % WGN;
    const int lm = lane & 15, lh = lane >> 4;

    auto issue = [&](int k0, int buf) {
        const unsigned base = (unsigned)(buf * BUFBYTES);
        #pragma unroll
        for (int c = t; c < TBM * (BK / 8); c += 256) {
            int row = c >> 3, col = (c & 7) * 8;
            unsigned lds = base + (unsigned)(row * LROW + col) * 2u;
            unsigned go  = (unsigned)(((long)(m0 + row) * lda + k0 + col) * 2);
            asm volatile("global_load_async_to_lds_b128 %0, %1, %2"
                         :: "v"(lds), "v"(go), "s"(A) : "memory");
        }
        #pragma unroll
        for (int c = t; c < TBN * (BK / 8); c += 256) {
            int row = c >> 3, col = (c & 7) * 8;
            unsigned lds = base + (unsigned)ABYTES + (unsigned)(row * LROW + col) * 2u;
            unsigned go  = (unsigned)(((long)(n0 + row) * ldb + k0 + col) * 2);
            asm volatile("global_load_async_to_lds_b128 %0, %1, %2"
                         :: "v"(lds), "v"(go), "s"(Bm) : "memory");
        }
    };

    v8f acc[FM][FN] = {};
    const int NS = Kdim / BK;

    issue(0, 0);
    for (int i = 0; i < NS; ++i) {
        const int buf = i & 1;
        if (i + 1 < NS) {
            issue((i + 1) * BK, (i + 1) & 1);
            asm volatile("s_wait_asynccnt %0" :: "n"(G) : "memory");  // stage i done
        } else {
            asm volatile("s_wait_asynccnt 0" ::: "memory");
        }
        __syncthreads();

        const bf16_t* sA = (const bf16_t*)(smem + buf * BUFBYTES);
        const bf16_t* sB = (const bf16_t*)(smem + buf * BUFBYTES + ABYTES);
        #pragma unroll
        for (int kc = 0; kc < BK; kc += 32) {
            v16bf af[FM], bf[FN];
            #pragma unroll
            for (int ii = 0; ii < FM; ++ii) af[ii] = load_afrag(sA, wm * WM + ii * 16, kc, lm, lh, LROW);
            #pragma unroll
            for (int j = 0; j < FN; ++j)    bf[j]  = load_bfrag(sB, wn * WN + j * 16,  kc, lm, lh, LROW);
            #pragma unroll
            for (int ii = 0; ii < FM; ++ii)
                #pragma unroll
                for (int j = 0; j < FN; ++j)
                    acc[ii][j] = wmma_bf16(af[ii], bf[j], acc[ii][j]);
        }
        __syncthreads();
    }

    const int om = m0 + wm * WM;
    const int on = n0 + wn * WN;
    if (OM == 0) {
        bf16_t* O = (bf16_t*)Ob + (long)z * oStr;
        #pragma unroll
        for (int i = 0; i < FM; ++i)
            #pragma unroll
            for (int j = 0; j < FN; ++j)
                #pragma unroll
                for (int v = 0; v < 8; ++v)
                    O[(long)(om + i * 16 + 8 * lh + v) * ldo + on + j * 16 + lm] = (bf16_t)acc[i][j][v];
    } else if (OM == 1) {
        bf16_t* O = (bf16_t*)Ob + (long)z * oStr;     // O[n][m]
        #pragma unroll
        for (int i = 0; i < FM; ++i)
            #pragma unroll
            for (int j = 0; j < FN; ++j) {
                v8bf pk;
                #pragma unroll
                for (int v = 0; v < 8; ++v) pk[v] = (bf16_t)acc[i][j][v];
                *(v8bf*)&O[(long)(on + j * 16 + lm) * ldo + om + i * 16 + 8 * lh] = pk;
            }
    } else {
        float* O = (float*)Ob + (long)z * oStr;
        #pragma unroll
        for (int i = 0; i < FM; ++i)
            #pragma unroll
            for (int j = 0; j < FN; ++j) {
                int n = on + j * 16 + lm;
                float bb = bias ? bias[n] : 0.0f;
                #pragma unroll
                for (int v = 0; v < 8; ++v)
                    O[(long)(om + i * 16 + 8 * lh + v) * ldo + n] = acc[i][j][v] + bb;
            }
    }
}

// ---------------------------------------------------------------- fused attention
// per (s-tile 64, head, batch): scores = Q*KpT^T/8 -> softmax(256) -> P*VpT^T.
// Tiles fetched by the Tensor Data Mover (one wave issues, TENSORcnt-synced).
__global__ __launch_bounds__(256) void linformer_attn(
    const bf16_t* __restrict__ Qg, const bf16_t* __restrict__ Kpg,
    const bf16_t* __restrict__ Vpg, bf16_t* __restrict__ Og)
{
    constexpr int S = 4096, H = 16, D = 64, KL = 256;
    constexpr int LQ = 72, LV = 264;                  // padded LDS row strides (halves)
    constexpr unsigned OFF_Q  = 0;
    constexpr unsigned OFF_KP = OFF_Q  + 64  * LQ * 2;  //  9216
    constexpr unsigned OFF_VP = OFF_KP + 256 * LQ * 2;  // 46080
    constexpr unsigned OFF_P  = OFF_VP + 64  * LV * 2;  // 79872
    constexpr unsigned OFF_RED = OFF_P + 64  * LV * 2;  // 113664

    extern __shared__ char smem[];
    bf16_t* sQ  = (bf16_t*)(smem + OFF_Q);
    bf16_t* sKp = (bf16_t*)(smem + OFF_KP);
    bf16_t* sVp = (bf16_t*)(smem + OFF_VP);
    bf16_t* sP  = (bf16_t*)(smem + OFF_P);
    float*  red = (float*)(smem + OFF_RED);           // [64][4]
    float*  rst = red + 64 * 4;                       // [64]

    const int s0 = blockIdx.x * 64;
    const int hh = blockIdx.y;
    const int b  = blockIdx.z;
    const int t = threadIdx.x, w = t >> 5, lane = t & 31;
    const int wm = w >> 2, wn = w & 3;                // 2 x 4 waves
    const int lm = lane & 15, lh = lane >> 4;

    const bf16_t* Qb  = Qg  + ((long)(b * H + hh) * S + s0) * D;
    const bf16_t* Kpb = Kpg + (long)(b * H + hh) * KL * D;
    const bf16_t* Vpb = Vpg + (long)(b * H + hh) * D * KL;

    // TDM ignores EXEC -> must be skipped by a *scalar* branch for waves 1..7
    if (__builtin_amdgcn_readfirstlane((int)threadIdx.x) < 32) {
        tdm_load_2d(OFF_Q,  Qb,  64, 64,  64,  4, 3);   // 128B rows, +16B pad
        tdm_load_2d(OFF_KP, Kpb, 64, 256, 64,  4, 3);
        tdm_load_2d(OFF_VP, Vpb, 256, 64, 256, 6, 3);   // 512B rows, +16B pad
    }
    __builtin_amdgcn_s_wait_tensorcnt(0);
    __syncthreads();

    // ---- scores: 64x256, each wave 32x64 (FM=2, FN=4), K = D = 64
    v8f acc[2][4] = {};
    #pragma unroll
    for (int kc = 0; kc < 64; kc += 32) {
        v16bf af[2], bf[4];
        #pragma unroll
        for (int i = 0; i < 2; ++i) af[i] = load_afrag(sQ,  wm * 32 + i * 16, kc, lm, lh, LQ);
        #pragma unroll
        for (int j = 0; j < 4; ++j) bf[j] = load_bfrag(sKp, wn * 64 + j * 16, kc, lm, lh, LQ);
        #pragma unroll
        for (int i = 0; i < 2; ++i)
            #pragma unroll
            for (int j = 0; j < 4; ++j) acc[i][j] = wmma_bf16(af[i], bf[j], acc[i][j]);
    }
    #pragma unroll
    for (int i = 0; i < 2; ++i)
        #pragma unroll
        for (int j = 0; j < 4; ++j)
            #pragma unroll
            for (int v = 0; v < 8; ++v) acc[i][j][v] *= 0.125f;   // 1/sqrt(64)

    // ---- softmax over 256 (row r = wm*32 + i*16 + 8*lh + v)
    float pmax[2][8];
    #pragma unroll
    for (int i = 0; i < 2; ++i)
        #pragma unroll
        for (int v = 0; v < 8; ++v) {
            float m = acc[i][0][v];
            #pragma unroll
            for (int j = 1; j < 4; ++j) m = fmaxf(m, acc[i][j][v]);
            #pragma unroll
            for (int off = 8; off >= 1; off >>= 1) m = fmaxf(m, __shfl_xor(m, off, 32));
            pmax[i][v] = m;
        }
    if (lm == 0) {
        #pragma unroll
        for (int i = 0; i < 2; ++i)
            #pragma unroll
            for (int v = 0; v < 8; ++v)
                red[(wm * 32 + i * 16 + 8 * lh + v) * 4 + wn] = pmax[i][v];
    }
    __syncthreads();
    if (t < 64) rst[t] = fmaxf(fmaxf(red[t*4], red[t*4+1]), fmaxf(red[t*4+2], red[t*4+3]));
    __syncthreads();

    float psum[2][8];
    #pragma unroll
    for (int i = 0; i < 2; ++i)
        #pragma unroll
        for (int v = 0; v < 8; ++v) {
            float rm = rst[wm * 32 + i * 16 + 8 * lh + v];
            float s = 0.0f;
            #pragma unroll
            for (int j = 0; j < 4; ++j) {
                float e = __expf(acc[i][j][v] - rm);
                acc[i][j][v] = e;
                s += e;
            }
            #pragma unroll
            for (int off = 8; off >= 1; off >>= 1) s += __shfl_xor(s, off, 32);
            psum[i][v] = s;
        }
    if (lm == 0) {
        #pragma unroll
        for (int i = 0; i < 2; ++i)
            #pragma unroll
            for (int v = 0; v < 8; ++v)
                red[(wm * 32 + i * 16 + 8 * lh + v) * 4 + wn] = psum[i][v];
    }
    __syncthreads();
    if (t < 64) rst[t] = 1.0f / (red[t*4] + red[t*4+1] + red[t*4+2] + red[t*4+3]);
    __syncthreads();

    #pragma unroll
    for (int i = 0; i < 2; ++i)
        #pragma unroll
        for (int v = 0; v < 8; ++v) {
            int r = wm * 32 + i * 16 + 8 * lh + v;
            float inv = rst[r];
            #pragma unroll
            for (int j = 0; j < 4; ++j)
                sP[r * LV + wn * 64 + j * 16 + lm] = (bf16_t)(acc[i][j][v] * inv);
        }
    __syncthreads();

    // ---- out = P(64x256) * VpT(64x256)^T : each wave 32x16, K = 256
    v8f oacc[2] = {};
    #pragma unroll
    for (int kc = 0; kc < 256; kc += 32) {
        v16bf af[2];
        #pragma unroll
        for (int i = 0; i < 2; ++i) af[i] = load_afrag(sP, wm * 32 + i * 16, kc, lm, lh, LV);
        v16bf bf = load_bfrag(sVp, wn * 16, kc, lm, lh, LV);
        #pragma unroll
        for (int i = 0; i < 2; ++i) oacc[i] = wmma_bf16(af[i], bf, oacc[i]);
    }

    #pragma unroll
    for (int i = 0; i < 2; ++i)
        #pragma unroll
        for (int v = 0; v < 8; ++v) {
            int s = s0 + wm * 32 + i * 16 + 8 * lh + v;
            int d = wn * 16 + lm;
            Og[((long)(b * S + s)) * (H * D) + hh * D + d] = (bf16_t)oacc[i][v];
        }
}

// ---------------------------------------------------------------- host
extern "C" void kernel_launch(void* const* d_in, const int* in_sizes, int n_in,
                              void* d_out, int out_size, void* d_ws, size_t ws_size,
                              hipStream_t stream) {
    (void)in_sizes; (void)n_in; (void)out_size; (void)ws_size;
    constexpr int B = 4, S = 4096, C = 1024, H = 16, D = 64, KL = 256;

    const float* x   = (const float*)d_in[0];
    const float* Wq  = (const float*)d_in[1];
    const float* Wk  = (const float*)d_in[2];
    const float* Wv  = (const float*)d_in[3];
    const float* We  = (const float*)d_in[4];
    const float* Wf  = (const float*)d_in[5];
    const float* Wo  = (const float*)d_in[6];
    const float* bo  = (const float*)d_in[7];

    char* ws = (char*)d_ws;
    size_t off = 0;
    auto carve = [&](size_t bytes) -> char* {
        char* p = ws + off;
        off = (off + bytes + 255) & ~(size_t)255;
        return p;
    };
    bf16_t* xh   = (bf16_t*)carve((size_t)B * S * C * 2);
    bf16_t* Wqh  = (bf16_t*)carve((size_t)H * D * C * 2);
    bf16_t* Wkh  = (bf16_t*)carve((size_t)H * D * C * 2);
    bf16_t* Wvh  = (bf16_t*)carve((size_t)H * D * C * 2);
    bf16_t* Weh  = (bf16_t*)carve((size_t)H * KL * S * 2);
    bf16_t* Wfh  = (bf16_t*)carve((size_t)H * KL * S * 2);
    bf16_t* Woh  = (bf16_t*)carve((size_t)C * H * D * 2);
    bf16_t* Qb   = (bf16_t*)carve((size_t)B * H * S * D * 2);
    bf16_t* KhT  = (bf16_t*)carve((size_t)B * H * D * S * 2);
    bf16_t* VhT  = (bf16_t*)carve((size_t)B * H * D * S * 2);
    bf16_t* KpT  = (bf16_t*)carve((size_t)B * H * KL * D * 2);
    bf16_t* VpT  = (bf16_t*)carve((size_t)B * H * D * KL * 2);
    bf16_t* out2 = (bf16_t*)carve((size_t)B * S * H * D * 2);

    cvt_f32_bf16<<<dim3(4096), dim3(256), 0, stream>>>(x,  xh,  (long)B * S * C);
    cvt_f32_bf16<<<dim3(1024), dim3(256), 0, stream>>>(Wq, Wqh, (long)H * D * C);
    cvt_f32_bf16<<<dim3(1024), dim3(256), 0, stream>>>(Wk, Wkh, (long)H * D * C);
    cvt_f32_bf16<<<dim3(1024), dim3(256), 0, stream>>>(Wv, Wvh, (long)H * D * C);
    cvt_f32_bf16<<<dim3(4096), dim3(256), 0, stream>>>(We, Weh, (long)H * KL * S);
    cvt_f32_bf16<<<dim3(4096), dim3(256), 0, stream>>>(Wf, Wfh, (long)H * KL * S);
    cvt_f32_bf16<<<dim3(1024), dim3(256), 0, stream>>>(Wo, Woh, (long)C * H * D);

    constexpr size_t SMEM_G = (size_t)2 * (128 + 64) * 72 * 2;   // double-buffered, 55,296 B

    // Q[b,h](S,D) = xh[b](S,C) * Wq[h](D,C)^T
    gemm_abt<128,64,4,2,0><<<dim3(1, S/128, B*H), 256, SMEM_G, stream>>>(
        xh, Wqh, Qb, nullptr, C, C, C, D,
        (long)S*C, (long)D*C, (long)S*D, 1, 2, H);
    // KhT[b,h](D,S) = transpose(xh[b] * Wk[h]^T)
    gemm_abt<128,64,4,2,1><<<dim3(1, S/128, B*H), 256, SMEM_G, stream>>>(
        xh, Wkh, KhT, nullptr, C, C, C, S,
        (long)S*C, (long)D*C, (long)D*S, 1, 2, H);
    // VhT[b,h](D,S)
    gemm_abt<128,64,4,2,1><<<dim3(1, S/128, B*H), 256, SMEM_G, stream>>>(
        xh, Wvh, VhT, nullptr, C, C, C, S,
        (long)S*C, (long)D*C, (long)D*S, 1, 2, H);
    // KpT[b,h](KL,D) = We[h](KL,S) * KhT[b,h](D,S)^T
    gemm_abt<128,64,4,2,0><<<dim3(1, KL/128, B*H), 256, SMEM_G, stream>>>(
        Weh, KhT, KpT, nullptr, S, S, S, D,
        (long)KL*S, (long)D*S, (long)KL*D, 2, 0, H);
    // VpT[b,h](D,KL) = VhT[b,h](D,S) * Wf[h](KL,S)^T
    gemm_abt<64,128,2,4,0><<<dim3(KL/128, 1, B*H), 256, SMEM_G, stream>>>(
        VhT, Wfh, VpT, nullptr, S, S, S, KL,
        (long)D*S, (long)KL*S, (long)D*KL, 0, 2, H);

    // fused scores -> softmax -> P*VpT^T, concat layout [B,S,H*D]
    constexpr size_t SMEM_A =
        ((size_t)64*72 + 256*72 + 64*264 + 64*264) * 2 + (64*4 + 64) * 4;  // 114,944 B
    linformer_attn<<<dim3(S/64, H, B), 256, SMEM_A, stream>>>(Qb, KpT, VpT, out2);

    // final: out[B*S,C] = out2[B*S,HD] * Wo[C,HD]^T + bo  (fp32 output)
    gemm_abt<128,64,4,2,2><<<dim3(C/64, (B*S)/128, 1), 256, SMEM_G, stream>>>(
        out2, Woh, d_out, bo, H*D, H*D, H*D, C,
        0L, 0L, 0L, 0, 0, H);
}
